// bi_model_x4_67860483277659
// MI455X (gfx1250) — compile-verified, matrix operand order checked by reading, not run on hardware
//
#include <hip/hip_runtime.h>
#include <hip/hip_bf16.h>
#include <math.h>

// ---------------- problem constants ----------------
#define Bq    2
#define Cc    180
#define Cp    192          // C padded to mult of 32 (WMMA K)
#define WSZ   8
#define SHF   4
#define NHd   6
#define HD    30
#define HDp   32
#define Nt    64           // tokens per window
#define NWI   1024         // windows per image (32*32)
#define NWIN  2048         // B * NWI
#define MR    131072       // B * H * W  (GEMM M)
#define HID   720
#define HIDp  736          // padded to mult of 32
#define QKN   540
#define QKNb  640          // qkv N padded to mult of 128
#define PRJb  256          // proj N padded to mult of 128
#define FC1b  768          // fc1 N padded to mult of 128
#define FC2b  256          // fc2 N padded to mult of 128
#define SCL   0.18257418583505536f   // 30^-0.5

typedef __bf16 bf16;
typedef __attribute__((ext_vector_type(16))) __bf16 v16bf;
typedef __attribute__((ext_vector_type(8)))  float  v8f;

// ---- WMMA fragment loader (ISA 7.12.2: 16-bit A 16x32 layout) ----
__device__ __forceinline__ v16bf load_frag(const bf16* base, int ld, int mbase, int kbase, int lane) {
  int m  = mbase + (lane & 15);
  int kh = (lane >> 4) << 3;
  const bf16* row = base + (size_t)m * ld + kbase + kh;
  v16bf f;
#pragma unroll
  for (int j = 0; j < 8; ++j) f[j] = row[j];
#pragma unroll
  for (int j = 0; j < 8; ++j) f[8 + j] = row[16 + j];
  return f;
}

__device__ __forceinline__ v8f wmma_bf16(v16bf a, v16bf b, v8f c) {
  return __builtin_amdgcn_wmma_f32_16x16x32_bf16(false, a, false, b, (short)0, c, false, false);
}

// ---- CDNA5 async copy global -> LDS (VGLOBAL op 98), tracked by ASYNCcnt ----
__device__ __forceinline__ void async_load_b128(const void* gptr, void* lptr) {
  unsigned lds_addr = (unsigned)(uintptr_t)lptr;     // LDS aperture: low 32 bits = DS offset
  unsigned long long ga = (unsigned long long)(uintptr_t)gptr;
  asm volatile("global_load_async_to_lds_b128 %0, %1, off"
               :: "v"(lds_addr), "v"(ga) : "memory");
}
__device__ __forceinline__ void wait_async0() {
  asm volatile("s_wait_asynccnt 0x0" ::: "memory");
}

__device__ __forceinline__ float gelu_exact(float v) {
  return 0.5f * v * (1.f + erff(v * 0.70710678118654752f));
}

// ---- tiled 128x128 block GEMM core: A [M x Kp], Bt [Npad x Kp], bf16, f32 accum ----
// 8 waves as 4(m) x 2(n); wave computes 32x64 => acc[2][4] 16x16 tiles.
// Double-buffered LDS staging via async-to-LDS, software pipelined one K-stage ahead.
__device__ __forceinline__ void gemm_block(const bf16* __restrict__ A, const bf16* __restrict__ Bt,
                                           int Kp, int mbase, int nbase, v8f acc[2][4]) {
  __shared__ bf16 As[2][128][32];
  __shared__ bf16 Bs[2][128][32];
  const int tid = threadIdx.x, lane = tid & 31, wave = tid >> 5;
  const int wm = (wave >> 1) * 32, wn = (wave & 1) * 64;

#pragma unroll
  for (int i = 0; i < 2; ++i)
#pragma unroll
    for (int j = 0; j < 4; ++j)
      acc[i][j] = (v8f){0.f,0.f,0.f,0.f,0.f,0.f,0.f,0.f};

  auto issue = [&](int buf, int kb) {
#pragma unroll
    for (int c = 0; c < 2; ++c) {
      int id  = tid + c * 256;          // 512 16B chunks per 128x32 tile
      int row = id >> 2, col = (id & 3) * 8;
      async_load_b128(A  + (size_t)(mbase + row) * Kp + kb + col, &As[buf][row][col]);
      async_load_b128(Bt + (size_t)(nbase + row) * Kp + kb + col, &Bs[buf][row][col]);
    }
  };

  issue(0, 0);
  for (int kb = 0; kb < Kp; kb += 32) {
    int buf = (kb >> 5) & 1;
    wait_async0();          // own wave's stage-kb loads landed
    __syncthreads();        // => whole tile resident; prior reads of buf^1 finished
    if (kb + 32 < Kp) issue(buf ^ 1, kb + 32);
    v16bf af[2], bfr[4];
#pragma unroll
    for (int i = 0; i < 2; ++i) af[i]  = load_frag(&As[buf][0][0], 32, wm + i * 16, 0, lane);
#pragma unroll
    for (int j = 0; j < 4; ++j) bfr[j] = load_frag(&Bs[buf][0][0], 32, wn + j * 16, 0, lane);
#pragma unroll
    for (int i = 0; i < 2; ++i)
#pragma unroll
      for (int j = 0; j < 4; ++j)
        acc[i][j] = wmma_bf16(af[i], bfr[j], acc[i][j]);
    __syncthreads();        // don't let fast waves re-stage over a buffer still being read
  }
}

// ---------------- K0: weight pack (transpose + zero-pad + bf16) ----------------
__global__ void wt_pack_kernel(const float* __restrict__ src, bf16* __restrict__ dst,
                               int Kreal, int Nreal, int Kp, int Np) {
  int i = blockIdx.x * 256 + threadIdx.x;
  if (i >= Np * Kp) return;
  int n = i / Kp, k = i % Kp;
  float v = (k < Kreal && n < Nreal) ? src[(size_t)k * Nreal + n] : 0.f;
  dst[(size_t)n * Kp + k] = (bf16)v;
}

// ---------------- K1/K5: LayerNorm (+optional shift+window-partition) -> bf16 padded ----
__global__ void ln_part_kernel(const float* __restrict__ src, const float* __restrict__ g,
                               const float* __restrict__ be, bf16* __restrict__ dst, int shifted) {
  int lane = threadIdx.x & 31, wave = threadIdx.x >> 5;
  int row = blockIdx.x * 8 + wave;
  size_t srow;
  if (shifted) {
    int bb = row >> 16, rem = row & 65535;
    int wi = rem >> 6, n = rem & 63;
    int wh = wi >> 5, wwc = wi & 31;
    int ph = wh * WSZ + (n >> 3), pw = wwc * WSZ + (n & 7);
    int h = (ph + SHF) & 255, w = (pw + SHF) & 255;
    srow = ((size_t)bb << 16) + (size_t)h * 256 + w;
  } else {
    srow = (size_t)row;
  }
  const float* p = src + srow * Cc;
  float s = 0.f, ss = 0.f;
  for (int c = lane; c < Cc; c += 32) { float v = p[c]; s += v; ss += v * v; }
  for (int o = 16; o > 0; o >>= 1) { s += __shfl_xor(s, o, 32); ss += __shfl_xor(ss, o, 32); }
  float mean = s * (1.f / Cc);
  float var  = ss * (1.f / Cc) - mean * mean;
  float rstd = rsqrtf(var + 1e-5f);
  bf16* q = dst + (size_t)row * Cp;
  for (int c = lane; c < Cp; c += 32) {
    float v = (c < Cc) ? ((p[c] - mean) * rstd * g[c] + be[c]) : 0.f;
    q[c] = (bf16)v;
  }
}

// ---------------- K2: QKV GEMM + bias + q-scale, scatter to (win,head) tiles ----------
__global__ void qkv_gemm_kernel(const bf16* __restrict__ A, const bf16* __restrict__ Bt,
                                const float* __restrict__ bias,
                                bf16* __restrict__ qb, bf16* __restrict__ kb, bf16* __restrict__ vb) {
  int lane = threadIdx.x & 31, wave = threadIdx.x >> 5;
  int mbase = blockIdx.y * 128, nbase = blockIdx.x * 128;
  int wm = (wave >> 1) * 32, wn = (wave & 1) * 64;
  v8f acc[2][4];
  gemm_block(A, Bt, Cp, mbase, nbase, acc);
#pragma unroll
  for (int j = 0; j < 4; ++j) {
    int n0 = nbase + wn + j * 16 + (lane & 15);
    if (n0 >= QKN) continue;
    int which = n0 / Cc, rem = n0 % Cc, head = rem / HD, d = rem % HD;
    float bv = bias[n0];
    float scale = (which == 0) ? SCL : 1.f;
    bf16* dst = (which == 0) ? qb : (which == 1) ? kb : vb;
#pragma unroll
    for (int i = 0; i < 2; ++i)
#pragma unroll
      for (int r = 0; r < 8; ++r) {
        int m = mbase + wm + i * 16 + r + ((lane >> 4) << 3);
        int win = m >> 6, n = m & 63;
        dst[((size_t)(win * NHd + head) * Nt + n) * HDp + d] = (bf16)((acc[i][j][r] + bv) * scale);
      }
  }
}

// ---------------- K3: window attention, fully in LDS, WMMA for S and O -----------------
__global__ void attn_kernel(const bf16* __restrict__ qb, const bf16* __restrict__ kb,
                            const bf16* __restrict__ vb, const float* __restrict__ rel_t,
                            bf16* __restrict__ attn_out) {
  __shared__ bf16 q_s[64][32];
  __shared__ bf16 k_s[64][32];
  __shared__ bf16 vt_s[32][64];
  __shared__ float S_s[64][65];
  __shared__ bf16 P_s[64][64];

  const int tid  = threadIdx.x;
  const int lane = tid & 31, wave = tid >> 5;
  const int win  = blockIdx.x;
  const int head = blockIdx.y;
  const size_t base = ((size_t)win * NHd + head) * Nt * HDp;

  for (int i = tid; i < 64 * 32; i += 256) {
    int n = i >> 5, d = i & 31;
    bf16 z = (bf16)0.f;
    bf16 qv = (d < HD) ? qb[base + (size_t)n * HDp + d] : z;
    bf16 kv = (d < HD) ? kb[base + (size_t)n * HDp + d] : z;
    bf16 vv = (d < HD) ? vb[base + (size_t)n * HDp + d] : z;
    q_s[n][d] = qv; k_s[n][d] = kv; vt_s[d][n] = vv;
  }
  __syncthreads();

  const int wi = win & (NWI - 1);
  const int wh = wi >> 5, wwc = wi & 31;

  for (int t = wave * 2; t < wave * 2 + 2; ++t) {
    int mi = t >> 2, ni = t & 3;
    v16bf a = load_frag(&q_s[0][0], 32, mi * 16, 0, lane);
    v16bf b = load_frag(&k_s[0][0], 32, ni * 16, 0, lane);
    v8f acc = {0.f,0.f,0.f,0.f,0.f,0.f,0.f,0.f};
    acc = wmma_bf16(a, b, acc);
    int n2 = ni * 16 + (lane & 15);
    int r2 = n2 >> 3, c2 = n2 & 7;
    int ph2 = wh * WSZ + r2, pw2 = wwc * WSZ + c2;
    int rid2 = (ph2 < 248 ? 0 : (ph2 < 252 ? 1 : 2)) * 3 + (pw2 < 248 ? 0 : (pw2 < 252 ? 1 : 2));
#pragma unroll
    for (int r = 0; r < 8; ++r) {
      int m = mi * 16 + r + ((lane >> 4) << 3);
      int r1 = m >> 3, c1 = m & 7;
      int rel = (r1 - r2 + 7) * 15 + (c1 - c2 + 7);
      float bias = rel_t[rel * NHd + head];
      int ph1 = wh * WSZ + r1, pw1 = wwc * WSZ + c1;
      int rid1 = (ph1 < 248 ? 0 : (ph1 < 252 ? 1 : 2)) * 3 + (pw1 < 248 ? 0 : (pw1 < 252 ? 1 : 2));
      float mv = (rid1 != rid2) ? -100.f : 0.f;
      S_s[m][n2] = acc[r] + bias + mv;
    }
  }
  __syncthreads();

  if (tid < 64) {
    float mx = -1e30f;
    for (int j = 0; j < 64; ++j) mx = fmaxf(mx, S_s[tid][j]);
    float sum = 0.f;
    for (int j = 0; j < 64; ++j) sum += __expf(S_s[tid][j] - mx);
    float inv = 1.f / sum;
    for (int j = 0; j < 64; ++j) P_s[tid][j] = (bf16)(__expf(S_s[tid][j] - mx) * inv);
  }
  __syncthreads();

  {
    int mi = wave >> 1, ni = wave & 1;
    v8f acc = {0.f,0.f,0.f,0.f,0.f,0.f,0.f,0.f};
    for (int kbv = 0; kbv < 64; kbv += 32) {
      v16bf a = load_frag(&P_s[0][0], 64, mi * 16, kbv, lane);
      v16bf b = load_frag(&vt_s[0][0], 64, ni * 16, kbv, lane);
      acc = wmma_bf16(a, b, acc);
    }
    int d = ni * 16 + (lane & 15);
    if (d < HD) {
#pragma unroll
      for (int r = 0; r < 8; ++r) {
        int m = mi * 16 + r + ((lane >> 4) << 3);
        attn_out[((size_t)win * Nt + m) * Cp + head * HD + d] = (bf16)acc[r];
      }
    }
  }
  if (head == 0) {
    for (int i = tid; i < 64 * (Cp - Cc); i += 256) {
      int r = i / (Cp - Cc), c = Cc + i % (Cp - Cc);
      attn_out[((size_t)win * Nt + r) * Cp + c] = (bf16)0.f;
    }
  }
}

// ---------------- K4: proj GEMM + window-reverse + un-shift + residual -> d_out (f32) ---
__global__ void proj_gemm_kernel(const bf16* __restrict__ A, const bf16* __restrict__ Bt,
                                 const float* __restrict__ bias, const float* __restrict__ xin,
                                 float* __restrict__ out) {
  int lane = threadIdx.x & 31, wave = threadIdx.x >> 5;
  int mbase = blockIdx.y * 128, nbase = blockIdx.x * 128;
  int wm = (wave >> 1) * 32, wn = (wave & 1) * 64;
  v8f acc[2][4];
  gemm_block(A, Bt, Cp, mbase, nbase, acc);
#pragma unroll
  for (int j = 0; j < 4; ++j) {
    int n0 = nbase + wn + j * 16 + (lane & 15);
    if (n0 >= Cc) continue;
    float bv = bias[n0];
#pragma unroll
    for (int i = 0; i < 2; ++i)
#pragma unroll
      for (int r = 0; r < 8; ++r) {
        int m = mbase + wm + i * 16 + r + ((lane >> 4) << 3);
        int bb = m >> 16, rem = m & 65535;
        int wi = rem >> 6, n = rem & 63;
        int wh = wi >> 5, wwc = wi & 31;
        int ph = wh * WSZ + (n >> 3), pw = wwc * WSZ + (n & 7);
        int h = (ph + SHF) & 255, w = (pw + SHF) & 255;
        size_t dst = (((size_t)bb << 16) + (size_t)h * 256 + w) * Cc + n0;
        out[dst] = xin[dst] + acc[i][j][r] + bv;
      }
  }
}

// ---------------- K6: FC1 GEMM + bias + GELU -> bf16 [MR x HIDp] -----------------------
__global__ void fc1_gemm_kernel(const bf16* __restrict__ A, const bf16* __restrict__ Bt,
                                const float* __restrict__ bias, bf16* __restrict__ h1) {
  int lane = threadIdx.x & 31, wave = threadIdx.x >> 5;
  int mbase = blockIdx.y * 128, nbase = blockIdx.x * 128;
  int wm = (wave >> 1) * 32, wn = (wave & 1) * 64;
  v8f acc[2][4];
  gemm_block(A, Bt, Cp, mbase, nbase, acc);
#pragma unroll
  for (int j = 0; j < 4; ++j) {
    int n0 = nbase + wn + j * 16 + (lane & 15);
    if (n0 >= HIDp) continue;                 // cols 720..735 store gelu(0)=0 (zero weights/bias)
    float bv = (n0 < HID) ? bias[n0] : 0.f;
#pragma unroll
    for (int i = 0; i < 2; ++i)
#pragma unroll
      for (int r = 0; r < 8; ++r) {
        int m = mbase + wm + i * 16 + r + ((lane >> 4) << 3);
        h1[(size_t)m * HIDp + n0] = (bf16)gelu_exact(acc[i][j][r] + bv);
      }
  }
}

// ---------------- K7: depthwise 3x3 conv + bias + GELU (bf16 in/out) -------------------
__global__ void dwconv_kernel(const bf16* __restrict__ h1, const float* __restrict__ w9,
                              const float* __restrict__ bi, bf16* __restrict__ h2) {
  int i = blockIdx.x * 256 + threadIdx.x;
  if (i >= Bq * 65536 * HIDp) return;
  int ch = i % HIDp;
  int l  = (i / HIDp) & 65535;
  int bb = i / (HIDp * 65536);
  size_t orow = ((size_t)bb << 16) + l;
  if (ch >= HID) { h2[orow * HIDp + ch] = (bf16)0.f; return; }
  int h = l >> 8, w = l & 255;
  float acc = bi[ch];
#pragma unroll
  for (int dh = -1; dh <= 1; ++dh) {
    int hh = h + dh; if (hh < 0 || hh > 255) continue;
#pragma unroll
    for (int dw = -1; dw <= 1; ++dw) {
      int ww2 = w + dw; if (ww2 < 0 || ww2 > 255) continue;
      acc += (float)h1[(((size_t)bb << 16) + (size_t)hh * 256 + ww2) * HIDp + ch]
             * w9[ch * 9 + (dh + 1) * 3 + (dw + 1)];
    }
  }
  h2[orow * HIDp + ch] = (bf16)gelu_exact(acc);
}

// ---------------- K8: FC2 GEMM + bias + residual into d_out ---------------------------
__global__ void fc2_gemm_kernel(const bf16* __restrict__ A, const bf16* __restrict__ Bt,
                                const float* __restrict__ bias, float* __restrict__ out) {
  int lane = threadIdx.x & 31, wave = threadIdx.x >> 5;
  int mbase = blockIdx.y * 128, nbase = blockIdx.x * 128;
  int wm = (wave >> 1) * 32, wn = (wave & 1) * 64;
  v8f acc[2][4];
  gemm_block(A, Bt, HIDp, mbase, nbase, acc);
#pragma unroll
  for (int j = 0; j < 4; ++j) {
    int n0 = nbase + wn + j * 16 + (lane & 15);
    if (n0 >= Cc) continue;
    float bv = bias[n0];
#pragma unroll
    for (int i = 0; i < 2; ++i)
#pragma unroll
      for (int r = 0; r < 8; ++r) {
        int m = mbase + wm + i * 16 + r + ((lane >> 4) << 3);
        size_t dst = (size_t)m * Cc + n0;
        out[dst] = out[dst] + acc[i][j][r] + bv;
      }
  }
}

// ---------------- host side ----------------
extern "C" void kernel_launch(void* const* d_in, const int* in_sizes, int n_in,
                              void* d_out, int out_size, void* d_ws, size_t ws_size,
                              hipStream_t stream) {
  const float* x      = (const float*)d_in[0];
  const float* n1g    = (const float*)d_in[1];
  const float* n1b    = (const float*)d_in[2];
  const float* qkv_w  = (const float*)d_in[3];
  const float* qkv_b  = (const float*)d_in[4];
  const float* rel_t  = (const float*)d_in[5];
  const float* proj_w = (const float*)d_in[6];
  const float* proj_b = (const float*)d_in[7];
  const float* n2g    = (const float*)d_in[8];
  const float* n2b    = (const float*)d_in[9];
  const float* fc1_w  = (const float*)d_in[10];
  const float* fc1_b  = (const float*)d_in[11];
  const float* dw_w   = (const float*)d_in[12];
  const float* dw_b   = (const float*)d_in[13];
  const float* fc2_w  = (const float*)d_in[14];
  const float* fc2_b  = (const float*)d_in[15];
  float* out = (float*)d_out;

  char* wsb = (char*)d_ws;
  size_t off = 0;
  auto alloc = [&](size_t bytes) -> char* {
    char* p = wsb + off;
    off = (off + bytes + 255) & ~(size_t)255;
    return p;
  };
  bf16* wT_qkv  = (bf16*)alloc((size_t)QKNb * Cp * 2);
  bf16* wT_proj = (bf16*)alloc((size_t)PRJb * Cp * 2);
  bf16* wT_fc1  = (bf16*)alloc((size_t)FC1b * Cp * 2);
  bf16* wT_fc2  = (bf16*)alloc((size_t)FC2b * HIDp * 2);

  const size_t QKV_ELEMS = (size_t)NWIN * NHd * Nt * HDp;
  const size_t XW_ELEMS  = (size_t)MR * Cp;
  char* slot1 = alloc(XW_ELEMS * 2);          // xw, later m2
  char* slot2 = alloc(4 * QKV_ELEMS * 2);     // q,k,v,attn_out; later h1
  char* slot3 = alloc((size_t)MR * HIDp * 2); // h2

  bf16* xw       = (bf16*)slot1;
  bf16* qbuf     = (bf16*)slot2;
  bf16* kbuf     = qbuf + QKV_ELEMS;
  bf16* vbuf     = kbuf + QKV_ELEMS;
  bf16* attn_out = vbuf + QKV_ELEMS;
  bf16* m2       = (bf16*)slot1;
  bf16* h1       = (bf16*)slot2;
  bf16* h2       = (bf16*)slot3;

  // K0: pack weights (transposed, padded, bf16)
  wt_pack_kernel<<<(QKNb * Cp + 255) / 256, 256, 0, stream>>>(qkv_w,  wT_qkv,  Cc, QKN, Cp, QKNb);
  wt_pack_kernel<<<(PRJb * Cp + 255) / 256, 256, 0, stream>>>(proj_w, wT_proj, Cc, Cc,  Cp, PRJb);
  wt_pack_kernel<<<(FC1b * Cp + 255) / 256, 256, 0, stream>>>(fc1_w,  wT_fc1,  Cc, HID, Cp, FC1b);
  wt_pack_kernel<<<(FC2b * HIDp + 255) / 256, 256, 0, stream>>>(fc2_w, wT_fc2, HID, Cc, HIDp, FC2b);

  // K1: LN1 + shift + window partition
  ln_part_kernel<<<MR / 8, 256, 0, stream>>>(x, n1g, n1b, xw, 1);

  // K2: QKV GEMM (M=131072, N=640pad, K=192)
  qkv_gemm_kernel<<<dim3(QKNb / 128, MR / 128), 256, 0, stream>>>(xw, wT_qkv, qkv_b, qbuf, kbuf, vbuf);

  // K3: windowed attention
  attn_kernel<<<dim3(NWIN, NHd), 256, 0, stream>>>(qbuf, kbuf, vbuf, rel_t, attn_out);

  // K4: proj GEMM + reverse shift + residual -> d_out (f32)
  proj_gemm_kernel<<<dim3(PRJb / 128, MR / 128), 256, 0, stream>>>(attn_out, wT_proj, proj_b, x, out);

  // K5: LN2 -> bf16
  ln_part_kernel<<<MR / 8, 256, 0, stream>>>(out, n2g, n2b, m2, 0);

  // K6: FC1 GEMM + GELU (N=768pad, K=192)
  fc1_gemm_kernel<<<dim3(FC1b / 128, MR / 128), 256, 0, stream>>>(m2, wT_fc1, fc1_b, h1);

  // K7: depthwise conv + GELU
  dwconv_kernel<<<(Bq * 65536 * HIDp + 255) / 256, 256, 0, stream>>>(h1, dw_w, dw_b, h2);

  // K8: FC2 GEMM + residual into d_out (N=256pad, K=736)
  fc2_gemm_kernel<<<dim3(FC2b / 128, MR / 128), 256, 0, stream>>>(h2, wT_fc2, fc2_b, out);
}
